// MultiheadAttention_26757646254568
// MI455X (gfx1250) — compile-verified, matrix operand order checked by reading, not run on hardware
//
#include <hip/hip_runtime.h>
#include <hip/hip_bf16.h>
#include <stddef.h>

// MI455X / gfx1250, wave32. All GEMM work on v_wmma_f32_16x16x32_f16.
// B=4, S=1024, F=512, H=16, D=32.

typedef _Float16 half8  __attribute__((ext_vector_type(8)));
typedef _Float16 half16 __attribute__((ext_vector_type(16)));
typedef float    float8 __attribute__((ext_vector_type(8)));
typedef float    float4v __attribute__((ext_vector_type(4)));
typedef int      int4v  __attribute__((ext_vector_type(4)));

#define Bb 4
#define Ss 1024
#define Ff 512
#define Hh 16
#define Dd 32
#define Mrows (Bb * Ss)          // 4096
#define RELROWS (2 * Ss + 1)     // 2049

#define GLOBAL_AS __attribute__((address_space(1)))
#define SHARED_AS __attribute__((address_space(3)))

// --- gfx1250 async global->LDS path (ASYNCcnt-tracked), with sync fallback ---
#if defined(__AMDGCN__) && __has_builtin(__builtin_amdgcn_global_load_async_to_lds_b128)
#define HAVE_ASYNC_LDS 1
#else
#define HAVE_ASYNC_LDS 0
#endif

__device__ __forceinline__ void async_copy16(const void* g, void* l) {
#if HAVE_ASYNC_LDS
  // builtin is typed by the 128-bit payload: (v4i global*, v4i lds*, imm, imm)
  __builtin_amdgcn_global_load_async_to_lds_b128(
      (GLOBAL_AS int4v*)g, (SHARED_AS int4v*)l, 0, 0);
#else
  *(half8*)l = *(const half8*)g;
#endif
}

__device__ __forceinline__ void async_wait0() {
#if HAVE_ASYNC_LDS
#if __has_builtin(__builtin_amdgcn_s_wait_asynccnt)
  __builtin_amdgcn_s_wait_asynccnt(0);
#else
  asm volatile("s_wait_asynccnt 0x0" ::: "memory");
#endif
#endif
}

__device__ __forceinline__ half16 combine(half8 lo, half8 hi) {
  half16 r;
#pragma unroll
  for (int i = 0; i < 8; ++i) { r[i] = lo[i]; r[i + 8] = hi[i]; }
  return r;
}

__device__ __forceinline__ half8 cvt8(const float* __restrict__ p) {
  float4v a = *(const float4v*)p;
  float4v b = *(const float4v*)(p + 4);
  half8 r;
  r[0] = (_Float16)a[0]; r[1] = (_Float16)a[1];
  r[2] = (_Float16)a[2]; r[3] = (_Float16)a[3];
  r[4] = (_Float16)b[0]; r[5] = (_Float16)b[1];
  r[6] = (_Float16)b[2]; r[7] = (_Float16)b[3];
  return r;
}

// ---------------------------------------------------------------------------
// R = rel_table @ Wrel  ->  f16 [2049][32]   (tiny: 4.2 MFLOP)
// ---------------------------------------------------------------------------
__global__ void relprep_kernel(const float* __restrict__ rel_table,
                               const float* __restrict__ Wrel,
                               _Float16* __restrict__ R) {
  int idx = blockIdx.x * 256 + threadIdx.x;
  if (idx >= RELROWS * Dd) return;
  int r = idx >> 5, c = idx & 31;
  float acc = 0.f;
#pragma unroll
  for (int d = 0; d < Dd; ++d)
    acc += rel_table[r * Dd + d] * Wrel[d * Dd + c];
  R[idx] = (_Float16)acc;
}

// ---------------------------------------------------------------------------
// Projection GEMM: C[4096x512] = A[4096x512](f32) @ W[512x512](f32)
// mode 0: write f16 to [b,h,s,d]   (Q,K attention layout)
// mode 1: write f16 to [b,h,d,s]   (V transposed, so PV B-operand is contiguous)
// mode 2: write f32 row-major      (final output)
// Block: 256 thr = 8 waves (2x4), WG tile 64(M) x 128(N), wave tile 32x32.
// W tile staged to LDS as f16 transposed [n][k] -> B fragments = 2x ds_load_b128.
// ---------------------------------------------------------------------------
__global__ __launch_bounds__(256) void proj_kernel(const float* __restrict__ A,
                                                   const float* __restrict__ W,
                                                   void* __restrict__ out,
                                                   int mode) {
  __shared__ _Float16 Wl[128 * 32];   // [n][k] transposed, 8 KB

  const int t = threadIdx.x;
  const int lane = t & 31, wid = t >> 5;
  const int lane15 = lane & 15, sel = lane >> 4;      // lane group 0/1
  const int wave_m = wid >> 2, wave_n = wid & 3;
  const int m0 = blockIdx.x * 64, n0 = blockIdx.y * 128;

  float8 c[2][2] = {};

  for (int kk = 0; kk < Ff; kk += 32) {
    __syncthreads();
    { // stage W[kk..kk+31][n0..n0+127] -> Wl[n][k] (f32 -> f16)
      int r = t >> 3;              // 0..31 (K row)
      int cb = (t & 7) * 16;       // 0..112 (N col base)
      const float* src = W + (size_t)(kk + r) * Ff + n0 + cb;
      if (kk + 32 < Ff)
        __builtin_prefetch(src + 32 * Ff, 0, 0);   // next K-slice of W
#pragma unroll
      for (int i = 0; i < 16; ++i)
        Wl[(cb + i) * 32 + r] = (_Float16)src[i];
    }
    __syncthreads();

    half16 afr[2];
#pragma unroll
    for (int a = 0; a < 2; ++a) {
      int m = m0 + wave_m * 32 + a * 16 + lane15;
      const float* ap = A + (size_t)m * Ff + kk + sel * 8;
      afr[a] = combine(cvt8(ap), cvt8(ap + 16));
    }
    half16 bfr[2];
#pragma unroll
    for (int bi = 0; bi < 2; ++bi) {
      int n = wave_n * 32 + bi * 16 + lane15;   // local col within WG tile
      const _Float16* bp = Wl + n * 32 + sel * 8;
      bfr[bi] = combine(*(const half8*)bp, *(const half8*)(bp + 16));
    }
#pragma unroll
    for (int a = 0; a < 2; ++a)
#pragma unroll
      for (int bi = 0; bi < 2; ++bi)
        c[a][bi] = __builtin_amdgcn_wmma_f32_16x16x32_f16(
            false, afr[a], false, bfr[bi], (short)0, c[a][bi], false, false);
  }

  // C fragment layout: lane holds M = v + 8*(lane/16), N = lane%16
#pragma unroll
  for (int a = 0; a < 2; ++a)
#pragma unroll
    for (int bi = 0; bi < 2; ++bi)
#pragma unroll
      for (int v = 0; v < 8; ++v) {
        int m = m0 + wave_m * 32 + a * 16 + v + 8 * sel;
        int n = n0 + wave_n * 32 + bi * 16 + lane15;
        float val = c[a][bi][v];
        if (mode == 2) {
          ((float*)out)[(size_t)m * Ff + n] = val;
        } else {
          int bb = m >> 10, s = m & (Ss - 1);
          int h = n >> 5, dd = n & (Dd - 1);
          size_t idx = (mode == 0)
              ? ((size_t)(bb * Hh + h) * Ss + s) * Dd + dd
              : ((size_t)(bb * Hh + h) * Dd + dd) * Ss + s;
          ((_Float16*)out)[idx] = (_Float16)val;
        }
      }
}

// ---------------------------------------------------------------------------
// Flash attention with relative-position band.
// grid = (S/128 q-tiles, B*H). 256 thr = 8 waves; each wave owns 16 q rows
// and the full 64-wide key tile -> softmax stats stay inside a wave
// (16-lane xor-shuffle reductions).
// K/V tiles double-buffered and streamed with GLOBAL_LOAD_ASYNC_TO_LDS_B128:
// tile i+1 is issued right after the barrier while WMMAs consume tile i;
// s_wait_asynccnt 0 at loop top. (Buffer being written was last read in the previous
// iteration, so the loop-top barrier makes the overwrite safe.)
// Rel band per wave per k-tile: T[16x80] = Qfrag @ R[base..base+79]^T
// (5 WMMAs), staged in LDS, gathered along j = kk - qi + 15.
// ---------------------------------------------------------------------------
__global__ __launch_bounds__(256) void attn_kernel(
    const _Float16* __restrict__ Qb, const _Float16* __restrict__ Kb,
    const _Float16* __restrict__ Vb, const _Float16* __restrict__ Rt,
    const int* __restrict__ mask, float* __restrict__ AO) {
  __shared__ _Float16 Kl[2][64 * 32];  // [buf][key][d]   8 KB
  __shared__ _Float16 Vl[2][32 * 64];  // [buf][d][key]   8 KB
  __shared__ float    Tl[8][16 * 80];  // per-wave band  40 KB
  __shared__ _Float16 Pl[8][16 * 64];  // per-wave P     16 KB

  const int t = threadIdx.x, lane = t & 31, w = t >> 5;
  const int lane15 = lane & 15, sel = lane >> 4;
  const int q0 = blockIdx.x * 128;
  const int bh = blockIdx.y;                 // b*16 + h
  const int b = bh >> 4, h = bh & 15;
  const int qw = q0 + w * 16;                // this wave's first q row

  // per-thread tile-copy coordinates (16 B per thread per tile)
  const int vdd = t >> 3, vcc = (t & 7) * 8;
  auto issue_tiles = [&](int k0, int buf) {
    const _Float16* ks = Kb + ((size_t)bh * Ss + k0) * Dd + t * 8;
    async_copy16(ks, &Kl[buf][t * 8]);
    const _Float16* vs = Vb + ((size_t)bh * Dd + vdd) * Ss + k0 + vcc;
    async_copy16(vs, &Vl[buf][vdd * 64 + vcc]);
  };

  // Q fragment: A 16x32, row = lane%16, K chunks per lane group
  const _Float16* qp = Qb + ((size_t)bh * Ss + qw + lane15) * Dd + sel * 8;
  const half16 qfrag = combine(*(const half8*)qp, *(const half8*)(qp + 16));

  float mrow[8], lrow[8];
#pragma unroll
  for (int v = 0; v < 8; ++v) { mrow[v] = -3.0e38f; lrow[v] = 0.f; }
  float8 o[2] = {};
  const float scale = 0.17677669529663687f;  // 1/sqrt(32)

  issue_tiles(0, 0);                     // prologue: stream first tile
  int buf = 0;

  for (int k0 = 0; k0 < Ss; k0 += 64) {
    async_wait0();                       // tile `buf` resident in LDS
    __syncthreads();                     // all waves done with buf^1
    if (k0 + 64 < Ss) issue_tiles(k0 + 64, buf ^ 1);   // stream next tile

    // --- QK^T: 16x64, D=32 = one WMMA K-step -> 4 WMMAs ---
    float8 sc[4];
#pragma unroll
    for (int n = 0; n < 4; ++n) {
      const _Float16* kp = &Kl[buf][(n * 16 + lane15) * Dd + sel * 8];
      half16 bfr = combine(*(const half8*)kp, *(const half8*)(kp + 16));
      float8 z = {};
      sc[n] = __builtin_amdgcn_wmma_f32_16x16x32_f16(
          false, qfrag, false, bfr, (short)0, z, false, false);
    }

    // --- rel band: T[16x80] = Q @ R[base..base+79]^T, 5 WMMAs ---
    const int base = k0 - qw - 15 + Ss;   // in [1,1969]; base+79 <= 2048
#pragma unroll
    for (int jt = 0; jt < 5; ++jt) {
      const _Float16* rp = Rt + (size_t)(base + jt * 16 + lane15) * Dd + sel * 8;
      half16 bfr = combine(*(const half8*)rp, *(const half8*)(rp + 16));
      float8 z = {};
      float8 tt = __builtin_amdgcn_wmma_f32_16x16x32_f16(
          false, qfrag, false, bfr, (short)0, z, false, false);
#pragma unroll
      for (int v = 0; v < 8; ++v)
        Tl[w][(v + 8 * sel) * 80 + jt * 16 + lane15] = tt[v];
    }

    // --- combine + scale + mask, row max ---
    float mx[8];
#pragma unroll
    for (int v = 0; v < 8; ++v) mx[v] = -3.0e38f;
#pragma unroll
    for (int n = 0; n < 4; ++n)
#pragma unroll
      for (int v = 0; v < 8; ++v) {
        int qi = v + 8 * sel;
        int kk = n * 16 + lane15;
        float val = (sc[n][v] + Tl[w][qi * 80 + kk - qi + 15]) * scale;
        int mv = mask[((size_t)b * Ss + qw + qi) * Ss + k0 + kk];
        val = (mv > 0) ? val : -3.0e38f;
        sc[n][v] = val;
        mx[v] = fmaxf(mx[v], val);
      }
#pragma unroll
    for (int v = 0; v < 8; ++v)
#pragma unroll
      for (int off = 1; off < 16; off <<= 1)
        mx[v] = fmaxf(mx[v], __shfl_xor(mx[v], off, 32));

    float alpha[8], mnew[8], rs[8];
#pragma unroll
    for (int v = 0; v < 8; ++v) {
      mnew[v] = fmaxf(mrow[v], mx[v]);
      alpha[v] = __expf(mrow[v] - mnew[v]);
      mrow[v] = mnew[v];
      rs[v] = 0.f;
    }
    // --- P = exp(sc - mnew), stage to per-wave LDS for A-fragment relayout ---
#pragma unroll
    for (int n = 0; n < 4; ++n)
#pragma unroll
      for (int v = 0; v < 8; ++v) {
        float p = __expf(sc[n][v] - mnew[v]);
        rs[v] += p;
        Pl[w][(v + 8 * sel) * 64 + n * 16 + lane15] = (_Float16)p;
      }
#pragma unroll
    for (int v = 0; v < 8; ++v) {
#pragma unroll
      for (int off = 1; off < 16; off <<= 1)
        rs[v] += __shfl_xor(rs[v], off, 32);
      lrow[v] = lrow[v] * alpha[v] + rs[v];
    }
#pragma unroll
    for (int np = 0; np < 2; ++np)
#pragma unroll
      for (int v = 0; v < 8; ++v) o[np][v] *= alpha[v];

    // --- O += P[16x64] @ V[64x32]: 4 WMMAs ---
    half16 af[2];
#pragma unroll
    for (int kc = 0; kc < 2; ++kc) {
      const _Float16* pp = &Pl[w][lane15 * 64 + kc * 32 + sel * 8];
      af[kc] = combine(*(const half8*)pp, *(const half8*)(pp + 16));
    }
#pragma unroll
    for (int np = 0; np < 2; ++np)
#pragma unroll
      for (int kc = 0; kc < 2; ++kc) {
        const _Float16* vp = &Vl[buf][(np * 16 + lane15) * 64 + kc * 32 + sel * 8];
        half16 bfr = combine(*(const half8*)vp, *(const half8*)(vp + 16));
        o[np] = __builtin_amdgcn_wmma_f32_16x16x32_f16(
            false, af[kc], false, bfr, (short)0, o[np], false, false);
      }
    buf ^= 1;
  }

  // epilogue: AO[(b*S+q)*F + h*32 + d] = O / l   (f32 for final GEMM)
#pragma unroll
  for (int np = 0; np < 2; ++np)
#pragma unroll
    for (int v = 0; v < 8; ++v) {
      int qi = v + 8 * sel;
      float l = lrow[v];
      float val = (l > 0.f) ? o[np][v] / l : 0.f;
      AO[((size_t)b * Ss + qw + qi) * Ff + h * Dd + np * 16 + lane15] = val;
    }
}

// ---------------------------------------------------------------------------
// launch: inputs (dict order): kv, q, mask, Wq, Wk, Wv, Wo, rel_table, Wrel
// ---------------------------------------------------------------------------
extern "C" void kernel_launch(void* const* d_in, const int* in_sizes, int n_in,
                              void* d_out, int out_size, void* d_ws, size_t ws_size,
                              hipStream_t stream) {
  const float* kv        = (const float*)d_in[0];
  const float* q         = (const float*)d_in[1];
  const int*   mask      = (const int*)d_in[2];
  const float* Wq        = (const float*)d_in[3];
  const float* Wk        = (const float*)d_in[4];
  const float* Wv        = (const float*)d_in[5];
  const float* Wo        = (const float*)d_in[6];
  const float* rel_table = (const float*)d_in[7];
  const float* Wrel      = (const float*)d_in[8];
  float* out = (float*)d_out;

  // workspace layout (16B-aligned slices)
  char* ws = (char*)d_ws;
  const size_t R_BYTES  = (size_t)RELROWS * Dd * sizeof(_Float16); // 131136, %16==0
  const size_t QKV_ELTS = (size_t)Bb * Hh * Ss * Dd;               // 2 MiB elems
  _Float16* R  = (_Float16*)ws;
  _Float16* Qb = (_Float16*)(ws + R_BYTES);
  _Float16* Kb = Qb + QKV_ELTS;
  _Float16* Vb = Kb + QKV_ELTS;
  float*    AO = (float*)(Vb + QKV_ELTS);                          // [4096][512] f32

  relprep_kernel<<<(RELROWS * Dd + 255) / 256, 256, 0, stream>>>(rel_table, Wrel, R);
  proj_kernel<<<dim3(Mrows / 64, Ff / 128), 256, 0, stream>>>(q,  Wq, (void*)Qb, 0);
  proj_kernel<<<dim3(Mrows / 64, Ff / 128), 256, 0, stream>>>(kv, Wk, (void*)Kb, 0);
  proj_kernel<<<dim3(Mrows / 64, Ff / 128), 256, 0, stream>>>(kv, Wv, (void*)Vb, 1);
  attn_kernel<<<dim3(Ss / 128, Bb * Hh), 256, 0, stream>>>(Qb, Kb, Vb, R, mask, AO);
  proj_kernel<<<dim3(Mrows / 64, Ff / 128), 256, 0, stream>>>(AO, Wo, (void*)out, 2);
}
